// PI_GNN_32452772888693
// MI455X (gfx1250) — matrix operand order repaired and checked by reading, not compiled, not run on hardware
//
#include <hip/hip_runtime.h>
#include <hip/hip_bf16.h>

// ---------------------------------------------------------------------------
// Problem constants (from reference)
// ---------------------------------------------------------------------------
#define BB   4
#define NPG  10000
#define EPG  32000
#define ML   128
#define NI   40000      // N = B*NPG
#define EI   128000     // E = B*EPG
#define E2I  16000      // EPG/2
#define ZETA 1e-6f
#define PEXP 1.852f
#define ILAYERS 3
#define KITERS  4
#define JHEADS  6

typedef __attribute__((ext_vector_type(16))) _Float16 v16h;
typedef __attribute__((ext_vector_type(8)))  _Float16 v8h;
typedef __attribute__((ext_vector_type(8)))  float    v8f;
typedef __attribute__((ext_vector_type(4)))  int      v4i;

// ---------------------------------------------------------------------------
// CDNA5 async global->LDS copy (ASYNCcnt-tracked), guarded by __has_builtin
// ---------------------------------------------------------------------------
#if __has_builtin(__builtin_amdgcn_global_load_async_to_lds_b128)
#define USE_ASYNC_LDS 1
#else
#define USE_ASYNC_LDS 0
#endif

// Builtin signature (from hipcc diagnostic): param0 is a non-const pointer to
// v4i in the global (__device__) address space; param1 is the LDS-side v4i*.
__device__ __forceinline__ __attribute__((address_space(1))) v4i*
glob_cast_v4(const void* p) {
  return (__attribute__((address_space(1))) v4i*)(unsigned long long)p;
}
__device__ __forceinline__ __attribute__((address_space(3))) v4i*
lds_cast_v4(void* p) {
  // AS3 pointers are 32-bit; low 32 bits of a generic LDS address are the
  // LDS byte offset (ISA 10.2 aperture rules), so integer round-trip is exact.
  return (__attribute__((address_space(3))) v4i*)(unsigned)(unsigned long long)p;
}

__device__ __forceinline__ void async_copy16(const _Float16* src, _Float16* dst) {
#if USE_ASYNC_LDS
  __builtin_amdgcn_global_load_async_to_lds_b128(glob_cast_v4(src),
                                                 lds_cast_v4(dst), 0, 0);
#else
  *(v8h*)dst = *(const v8h*)src;
#endif
}
__device__ __forceinline__ void async_wait() {
#if USE_ASYNC_LDS
#if __has_builtin(__builtin_amdgcn_s_wait_asynccnt)
  __builtin_amdgcn_s_wait_asynccnt(0);
#else
  asm volatile("s_wait_asynccnt 0" ::: "memory");
#endif
#endif
}

// ---------------------------------------------------------------------------
// Helpers
// ---------------------------------------------------------------------------
__device__ __forceinline__ float selu_f(float x) {
  const float a = 1.6732632423543772f, s = 1.0507009873554805f;
  return s * (x > 0.f ? x : a * (__expf(x) - 1.f));
}
// order-preserving float -> u32 key (monotone); key 0 reserved as "-inf/none"
__device__ __forceinline__ unsigned fkey(float f) {
  unsigned u = __float_as_uint(f);
  return (u & 0x80000000u) ? ~u : (u | 0x80000000u);
}
__device__ __forceinline__ float funkey(unsigned k) {
  unsigned u = (k & 0x80000000u) ? (k ^ 0x80000000u) : ~k;
  return __uint_as_float(u);
}

// A fragment: 16x32 f16 tile from LDS row-major [16 x lda], per ISA layout:
// lane L: row = L&15, K-base = (L>>4)*8; halves at +0..7 and +16..23.
__device__ __forceinline__ v16h load_a_frag(const _Float16* A, int lda, int k0) {
  const int lane = threadIdx.x & 31;
  const int row  = lane & 15;
  const int koff = (lane >> 4) * 8;
  const _Float16* p = A + row * lda + k0 + koff;
  v8h lo = *(const v8h*)(p);
  v8h hi = *(const v8h*)(p + 16);
  v16h a;
#pragma unroll
  for (int i = 0; i < 8; ++i) { a[i] = lo[i]; a[8 + i] = hi[i]; }
  return a;
}

// B fragment: 32x16 f16 from global weights row-major [K x ML]:
// lane L holds N column n0+(L&15), K rows (L>>4)*8+0..7 and +16..23.
__device__ __forceinline__ v16h load_b_frag(const _Float16* W, int k0, int n0) {
  const int lane = threadIdx.x & 31;
  const _Float16* p = W + (size_t)(k0 + (lane >> 4) * 8) * ML + n0 + (lane & 15);
  v16h b;
#pragma unroll
  for (int i = 0; i < 8; ++i) { b[i] = p[(size_t)i * ML]; b[8 + i] = p[(size_t)(16 + i) * ML]; }
  return b;
}

// One wave: 16 rows (A in LDS) x full 128 cols, K = KTILES*32.
template <int KTILES>
__device__ __forceinline__ void wave_gemm(const _Float16* Alds, int lda,
                                          const _Float16* Wg, v8f acc[8]) {
#pragma unroll
  for (int kt = 0; kt < KTILES; ++kt) {
    v16h a = load_a_frag(Alds, lda, kt * 32);
    if (kt + 1 < KTILES)
      __builtin_prefetch(Wg + (size_t)(kt * 32 + 32) * ML, 0, 1);
#pragma unroll
    for (int nt = 0; nt < 8; ++nt) {
      v16h b = load_b_frag(Wg, kt * 32, nt * 16);
      acc[nt] = __builtin_amdgcn_wmma_f32_16x16x32_f16(
          false, a, false, b, (short)0, acc[nt], false, false);
    }
  }
}

// Store 16x128 f32 accumulators to LDS as f16 (optionally through SELU).
__device__ __forceinline__ void store_tile_lds(_Float16* dst, int ldd,
                                               const v8f acc[8], bool do_selu) {
  const int lane = threadIdx.x & 31;
  const int n = lane & 15, hi = lane >> 4;
#pragma unroll
  for (int nt = 0; nt < 8; ++nt)
#pragma unroll
    for (int j = 0; j < 8; ++j) {
      float v = acc[nt][j];
      if (do_selu) v = selu_f(v);
      dst[(j + 8 * hi) * ldd + nt * 16 + n] = (_Float16)v;
    }
}

// ---------------------------------------------------------------------------
// NOTE on activation placement: g and z are only consumed through selu(.), and
// segment_max commutes with the monotone selu. So all producers store
// selu-activated f16 values, and the gather phases below are PURE COPIES
// (async global->LDS eligible). agg keys are over selu(z); empty segment -> 0
// (== selu(0)), matching the reference's where(agg>-1e30, agg, 0) under selu.
// ---------------------------------------------------------------------------

// Edge message layer: m=[gs[s],gs[r],zs]; z'=selu(m@We1)@We2; store selu(z'),
// segment_max(selu(z')) into aggk.
__global__ __launch_bounds__(128) void edge_layer_kernel(
    const _Float16* __restrict__ g, _Float16* __restrict__ z,
    const int* __restrict__ sndr, const int* __restrict__ rcvr,
    const _Float16* __restrict__ We1, const _Float16* __restrict__ We2,
    unsigned* __restrict__ aggk) {
  __shared__ _Float16 mt[64 * 392];   // 64 rows x 384 (+8 pad)
  __shared__ int sd[64], rc[64];
  const int tid = threadIdx.x;
  const int e0 = blockIdx.x * 64;

  if (tid < 64) { sd[tid] = sndr[e0 + tid]; rc[tid] = rcvr[e0 + tid]; }
  __syncthreads();

  for (int idx = tid * 8; idx < 64 * 384; idx += 128 * 8) {
    const int row = idx / 384, col = idx - row * 384;
    const _Float16* src;
    if (col < 128)      src = g + (size_t)sd[row] * ML + col;
    else if (col < 256) src = g + (size_t)rc[row] * ML + (col - 128);
    else                src = z + (size_t)(e0 + row) * ML + (col - 256);
    async_copy16(src, &mt[row * 392 + col]);
  }
  async_wait();
  __syncthreads();

  const int w = tid >> 5;
  const v8f zf = {0.f, 0.f, 0.f, 0.f, 0.f, 0.f, 0.f, 0.f};
  v8f acc[8];
#pragma unroll
  for (int i = 0; i < 8; ++i) acc[i] = zf;
  wave_gemm<12>(&mt[w * 16 * 392], 392, We1, acc);

  // overlay wave-private h1 (16x136) into this wave's own mt region
  _Float16* h1w = &mt[w * 16 * 392];
  store_tile_lds(h1w, 136, acc, true);

  v8f acc2[8];
#pragma unroll
  for (int i = 0; i < 8; ++i) acc2[i] = zf;
  wave_gemm<4>(h1w, 136, We2, acc2);

  const int lane = tid & 31, n = lane & 15, hi = lane >> 4;
#pragma unroll
  for (int nt = 0; nt < 8; ++nt)
#pragma unroll
    for (int j = 0; j < 8; ++j) {
      const int rloc = w * 16 + j + 8 * hi;
      const int col = nt * 16 + n;
      const float v = selu_f(acc2[nt][j]);   // activated store
      z[(size_t)(e0 + rloc) * ML + col] = (_Float16)v;
      atomicMax(&aggk[(size_t)rc[rloc] * ML + col], fkey(v));
    }
}

// Node update: g = selu( selu-domain [gs, agg] @ Wn1 ... ) -> store selu(g)
__global__ __launch_bounds__(128) void node_layer_kernel(
    _Float16* __restrict__ g, const unsigned* __restrict__ aggk,
    const _Float16* __restrict__ Wn1, const _Float16* __restrict__ Wn2) {
  __shared__ _Float16 mt[64 * 264];   // 64 rows x 256 (+8 pad)
  const int tid = threadIdx.x;
  const int n0 = blockIdx.x * 64;

  // first half: pure async copy of activated g rows
  for (int idx = tid * 8; idx < 64 * 128; idx += 128 * 8) {
    const int row = idx >> 7, col = idx & 127;
    async_copy16(g + (size_t)(n0 + row) * ML + col, &mt[row * 264 + col]);
  }
  // second half: decode segment_max keys (already selu-domain)
  for (int idx = tid * 8; idx < 64 * 128; idx += 128 * 8) {
    const int row = idx >> 7, col = idx & 127;
    const unsigned* kp = aggk + (size_t)(n0 + row) * ML + col;
    v8h o;
#pragma unroll
    for (int j = 0; j < 8; ++j) {
      unsigned k = kp[j];
      o[j] = (_Float16)(k ? funkey(k) : 0.f);  // empty -> selu(0) == 0
    }
    *(v8h*)&mt[row * 264 + 128 + col] = o;
  }
  async_wait();
  __syncthreads();

  const int w = tid >> 5;
  const v8f zf = {0.f, 0.f, 0.f, 0.f, 0.f, 0.f, 0.f, 0.f};
  v8f acc[8];
#pragma unroll
  for (int i = 0; i < 8; ++i) acc[i] = zf;
  wave_gemm<8>(&mt[w * 16 * 264], 264, Wn1, acc);

  _Float16* h1w = &mt[w * 16 * 264];
  store_tile_lds(h1w, 136, acc, true);

  v8f acc2[8];
#pragma unroll
  for (int i = 0; i < 8; ++i) acc2[i] = zf;
  wave_gemm<4>(h1w, 136, Wn2, acc2);

  const int lane = tid & 31, n = lane & 15, hi = lane >> 4;
#pragma unroll
  for (int nt = 0; nt < 8; ++nt)
#pragma unroll
    for (int j = 0; j < 8; ++j) {
      const int rloc = w * 16 + j + 8 * hi;
      g[(size_t)(n0 + rloc) * ML + nt * 16 + n] =
          (_Float16)selu_f(acc2[nt][j]);     // activated store
    }
}

// Flow head: lam=[gs[s],gs[r],zs]; dq=selu(selu(lam@Wf1)@Wf2)@Wf3; qtmp=q+dq
__global__ __launch_bounds__(128) void flow_kernel(
    const _Float16* __restrict__ g, const _Float16* __restrict__ z,
    const int* __restrict__ sndr, const int* __restrict__ rcvr,
    const _Float16* __restrict__ Wf1, const _Float16* __restrict__ Wf2,
    const float* __restrict__ Wf3, const float* __restrict__ qhat,
    float* __restrict__ qtmp) {
  __shared__ _Float16 mt[64 * 392];
  __shared__ int sd[64], rc[64];
  __shared__ float dsum[64];
  const int tid = threadIdx.x;
  const int e0 = blockIdx.x * 64;

  if (tid < 64) {
    sd[tid] = sndr[e0 + tid]; rc[tid] = rcvr[e0 + tid]; dsum[tid] = 0.f;
  }
  __syncthreads();

  for (int idx = tid * 8; idx < 64 * 384; idx += 128 * 8) {
    const int row = idx / 384, col = idx - row * 384;
    const _Float16* src;
    if (col < 128)      src = g + (size_t)sd[row] * ML + col;
    else if (col < 256) src = g + (size_t)rc[row] * ML + (col - 128);
    else                src = z + (size_t)(e0 + row) * ML + (col - 256);
    async_copy16(src, &mt[row * 392 + col]);
  }
  async_wait();
  __syncthreads();

  const int w = tid >> 5;
  const v8f zf = {0.f, 0.f, 0.f, 0.f, 0.f, 0.f, 0.f, 0.f};
  v8f acc[8];
#pragma unroll
  for (int i = 0; i < 8; ++i) acc[i] = zf;
  wave_gemm<12>(&mt[w * 16 * 392], 392, Wf1, acc);

  _Float16* h1w = &mt[w * 16 * 392];
  store_tile_lds(h1w, 136, acc, true);

  v8f acc2[8];
#pragma unroll
  for (int i = 0; i < 8; ++i) acc2[i] = zf;
  wave_gemm<4>(h1w, 136, Wf2, acc2);

  const int lane = tid & 31, n = lane & 15, hi = lane >> 4;
#pragma unroll
  for (int nt = 0; nt < 8; ++nt) {
    const float wv = Wf3[nt * 16 + n];
#pragma unroll
    for (int j = 0; j < 8; ++j)
      atomicAdd(&dsum[w * 16 + j + 8 * hi], selu_f(acc2[nt][j]) * wv);
  }
  __syncthreads();
  if (tid < 64) qtmp[e0 + tid] = qhat[e0 + tid] + dsum[tid];
}

// ---------------------------------------------------------------------------
// Elementwise / scatter kernels
// ---------------------------------------------------------------------------
__global__ void cvt_f16_kernel(const float* __restrict__ s,
                               _Float16* __restrict__ d, int n) {
  for (int i = blockIdx.x * blockDim.x + threadIdx.x; i < n;
       i += gridDim.x * blockDim.x)
    d[i] = (_Float16)s[i];
}
__global__ void zero_u32_kernel(unsigned* __restrict__ p, long n) {
  for (long i = blockIdx.x * (long)blockDim.x + threadIdx.x; i < n;
       i += (long)gridDim.x * blockDim.x)
    p[i] = 0u;
}
__global__ void copy_f32_kernel(const float* __restrict__ s,
                                float* __restrict__ d, int n) {
  for (int i = blockIdx.x * blockDim.x + threadIdx.x; i < n;
       i += gridDim.x * blockDim.x)
    d[i] = s[i];
}
__global__ void extract_kernel(const float* __restrict__ x,
                               float* __restrict__ hstar,
                               float* __restrict__ dstar) {
  int n = blockIdx.x * blockDim.x + threadIdx.x;
  if (n < NI) { hstar[n] = x[2 * n]; dstar[n] = x[2 * n + 1]; }
}
__global__ void netflow_kernel(const float* __restrict__ h,
                               const float* __restrict__ r,
                               const int* __restrict__ sndr,
                               const int* __restrict__ rcvr,
                               float* __restrict__ q, float* __restrict__ d,
                               int addd) {
  int e = blockIdx.x * blockDim.x + threadIdx.x;
  if (e >= EI) return;
  float dh = h[sndr[e]] - h[rcvr[e]];
  float s = (dh > 0.f) ? 1.f : ((dh < 0.f) ? -1.f : 0.f);
  float qv = s * powf((fabsf(dh) + ZETA) / (r[e] + ZETA), 1.0f / PEXP);
  q[e] = qv;
  if (addd) atomicAdd(&d[rcvr[e]], qv);
}
__global__ void segsum_kernel(const float* __restrict__ q,
                              const int* __restrict__ rcvr,
                              float* __restrict__ d) {
  int e = blockIdx.x * blockDim.x + threadIdx.x;
  if (e < EI) atomicAdd(&d[rcvr[e]], q[e]);
}
__global__ void antisym_kernel(const float* __restrict__ qtmp,
                               float* __restrict__ qhat) {
  int e = blockIdx.x * blockDim.x + threadIdx.x;
  if (e >= EI) return;
  int b = e / EPG, loc = e - b * EPG;
  qhat[e] = (loc < E2I) ? qtmp[e] : -qtmp[e - E2I];
}
__global__ void hl_kernel(const float* __restrict__ qhat,
                          const float* __restrict__ r,
                          float* __restrict__ hl) {
  int e = blockIdx.x * blockDim.x + threadIdx.x;
  if (e >= EI) return;
  float qv = qhat[e];
  float s = (qv > 0.f) ? 1.f : ((qv < 0.f) ? -1.f : 0.f);
  hl[e] = r[e] * s * powf(fabsf(qv), PEXP);
}
__global__ void head_edge_kernel(const float* __restrict__ h,
                                 const float* __restrict__ hl,
                                 const int* __restrict__ sndr,
                                 const int* __restrict__ rcvr,
                                 unsigned* __restrict__ aggh) {
  int e = blockIdx.x * blockDim.x + threadIdx.x;
  if (e >= EI) return;
  float cand = h[sndr[e]] - hl[e];
  atomicMax(&aggh[rcvr[e]], fkey(cand));
}
__global__ void head_node_kernel(const float* __restrict__ hstar,
                                 const unsigned* __restrict__ aggh,
                                 float* __restrict__ h) {
  int n = blockIdx.x * blockDim.x + threadIdx.x;
  if (n >= NI) return;
  unsigned k = aggh[n];
  float hc = h[n];
  float agg = k ? funkey(k) : hc;
  float hs = hstar[n];
  h[n] = (hs != 0.f) ? hs : fmaxf(hc, agg);
}
__global__ void ginit_kernel(const float* __restrict__ dhat,
                             const float* __restrict__ dstar,
                             const float* __restrict__ Wni,
                             _Float16* __restrict__ g) {
  int n = blockIdx.x * blockDim.x + threadIdx.x;
  if (n >= NI) return;
  float a = selu_f(dhat[n]), b = selu_f(dstar[n]);
#pragma unroll 4
  for (int c = 0; c < ML; ++c)
    g[(size_t)n * ML + c] = (_Float16)selu_f(a * Wni[c] + b * Wni[ML + c]);
}
__global__ void zinit_kernel(const float* __restrict__ qtil,
                             const float* __restrict__ qhat,
                             const float* __restrict__ Wei,
                             _Float16* __restrict__ z) {
  int e = blockIdx.x * blockDim.x + threadIdx.x;
  if (e >= EI) return;
  float a = selu_f(qtil[e]), b = selu_f(qhat[e]);
#pragma unroll 4
  for (int c = 0; c < ML; ++c)
    z[(size_t)e * ML + c] = (_Float16)selu_f(a * Wei[c] + b * Wei[ML + c]);
}

// ---------------------------------------------------------------------------
// Host orchestration
// ---------------------------------------------------------------------------
extern "C" void kernel_launch(void* const* d_in, const int* in_sizes, int n_in,
                              void* d_out, int out_size, void* d_ws,
                              size_t ws_size, hipStream_t stream) {
  (void)in_sizes; (void)n_in; (void)out_size; (void)ws_size;
  const float* x    = (const float*)d_in[0];
  const float* r    = (const float*)d_in[1];
  const float* Wni  = (const float*)d_in[2];
  const float* Wei  = (const float*)d_in[3];
  const float* Wf1  = (const float*)d_in[4];
  const float* Wf2  = (const float*)d_in[5];
  const float* Wf3  = (const float*)d_in[6];
  const float* We1  = (const float*)d_in[7];
  const float* We2  = (const float*)d_in[8];
  const float* Wn1  = (const float*)d_in[9];
  const float* Wn2  = (const float*)d_in[10];
  const int* sndr   = (const int*)d_in[11];
  const int* rcvr   = (const int*)d_in[12];
  float* out = (float*)d_out;

  char* p = (char*)d_ws;
  auto alloc = [&](size_t bytes) -> char* {
    char* q = p;
    p += (bytes + 255) & ~(size_t)255;
    return q;
  };
  _Float16* g    = (_Float16*)alloc((size_t)NI * ML * 2);
  _Float16* z    = (_Float16*)alloc((size_t)EI * ML * 2);
  unsigned* aggk = (unsigned*)alloc((size_t)NI * ML * 4);
  float* hstar = (float*)alloc((size_t)NI * 4);
  float* dstar = (float*)alloc((size_t)NI * 4);
  float* h     = (float*)alloc((size_t)NI * 4);
  float* dhat  = (float*)alloc((size_t)NI * 4);
  unsigned* aggh = (unsigned*)alloc((size_t)NI * 4);
  float* qhat = (float*)alloc((size_t)EI * 4);
  float* qtil = (float*)alloc((size_t)EI * 4);
  float* qtmp = (float*)alloc((size_t)EI * 4);
  float* hl   = (float*)alloc((size_t)EI * 4);
  _Float16* wf1h = (_Float16*)alloc((size_t)384 * ML * 2);
  _Float16* wf2h = (_Float16*)alloc((size_t)ML * ML * 2);
  _Float16* we1h = (_Float16*)alloc((size_t)ILAYERS * 384 * ML * 2);
  _Float16* we2h = (_Float16*)alloc((size_t)ILAYERS * ML * ML * 2);
  _Float16* wn1h = (_Float16*)alloc((size_t)ILAYERS * 256 * ML * 2);
  _Float16* wn2h = (_Float16*)alloc((size_t)ILAYERS * ML * ML * 2);

  const dim3 T256(256);
  auto gN  = dim3((NI + 255) / 256);
  auto gE  = dim3((EI + 255) / 256);

  // weight conversion f32 -> f16
  cvt_f16_kernel<<<dim3(96), T256, 0, stream>>>(Wf1, wf1h, 384 * ML);
  cvt_f16_kernel<<<dim3(64), T256, 0, stream>>>(Wf2, wf2h, ML * ML);
  cvt_f16_kernel<<<dim3(256), T256, 0, stream>>>(We1, we1h, ILAYERS * 384 * ML);
  cvt_f16_kernel<<<dim3(128), T256, 0, stream>>>(We2, we2h, ILAYERS * ML * ML);
  cvt_f16_kernel<<<dim3(192), T256, 0, stream>>>(Wn1, wn1h, ILAYERS * 256 * ML);
  cvt_f16_kernel<<<dim3(128), T256, 0, stream>>>(Wn2, wn2h, ILAYERS * ML * ML);

  extract_kernel<<<gN, T256, 0, stream>>>(x, hstar, dstar);
  zero_u32_kernel<<<gN, T256, 0, stream>>>((unsigned*)dhat, NI);
  netflow_kernel<<<gE, T256, 0, stream>>>(hstar, r, sndr, rcvr, qhat, dhat, 1);
  copy_f32_kernel<<<gE, T256, 0, stream>>>(qhat, qtil, EI);
  copy_f32_kernel<<<gN, T256, 0, stream>>>(hstar, h, NI);

  const dim3 TW(128);
  const dim3 gEdge(EI / 64);   // 2000 tiles
  const dim3 gNode(NI / 64);   // 625 tiles

  for (int k = 0; k < KITERS; ++k) {
    ginit_kernel<<<gN, T256, 0, stream>>>(dhat, dstar, Wni, g);
    zinit_kernel<<<gE, T256, 0, stream>>>(qtil, qhat, Wei, z);

    for (int i = 0; i < ILAYERS; ++i) {
      zero_u32_kernel<<<dim3(4096), T256, 0, stream>>>(aggk, (long)NI * ML);
      edge_layer_kernel<<<gEdge, TW, 0, stream>>>(
          g, z, sndr, rcvr, we1h + (size_t)i * 384 * ML,
          we2h + (size_t)i * ML * ML, aggk);
      node_layer_kernel<<<gNode, TW, 0, stream>>>(
          g, aggk, wn1h + (size_t)i * 256 * ML, wn2h + (size_t)i * ML * ML);
    }

    flow_kernel<<<gEdge, TW, 0, stream>>>(g, z, sndr, rcvr, wf1h, wf2h, Wf3,
                                          qhat, qtmp);
    antisym_kernel<<<gE, T256, 0, stream>>>(qtmp, qhat);

    zero_u32_kernel<<<gN, T256, 0, stream>>>((unsigned*)dhat, NI);
    segsum_kernel<<<gE, T256, 0, stream>>>(qhat, rcvr, dhat);
    hl_kernel<<<gE, T256, 0, stream>>>(qhat, r, hl);

    copy_f32_kernel<<<gN, T256, 0, stream>>>(hstar, h, NI);
    for (int j = 0; j < JHEADS; ++j) {
      zero_u32_kernel<<<gN, T256, 0, stream>>>(aggh, NI);
      head_edge_kernel<<<gE, T256, 0, stream>>>(h, hl, sndr, rcvr, aggh);
      head_node_kernel<<<gN, T256, 0, stream>>>(hstar, aggh, h);
    }
    netflow_kernel<<<gE, T256, 0, stream>>>(h, r, sndr, rcvr, qtil, dhat, 0);
  }

  copy_f32_kernel<<<gN, T256, 0, stream>>>(h, out, NI);
}
